// FastAutoencoder_37340445671558
// MI455X (gfx1250) — matrix operand order repaired and checked by reading, not compiled, not run on hardware
//
#include <hip/hip_runtime.h>
#include <hip/hip_bf16.h>

typedef __bf16 bf16_t;
typedef __bf16 v16bf __attribute__((ext_vector_type(16)));
typedef __bf16 v8bf  __attribute__((ext_vector_type(8)));
typedef float  v8f   __attribute__((ext_vector_type(8)));
typedef float  v4f   __attribute__((ext_vector_type(4)));
typedef int    v4i_vec __attribute__((vector_size(16)));   // matches builtin param pointee

#define KTOP   64
#define KSEL   256
#define THETA  0.1f
#define DEADTH 266

#if defined(__AMDGCN__) && \
    __has_builtin(__builtin_amdgcn_global_load_async_to_lds_b128) && \
    __has_builtin(__builtin_amdgcn_s_wait_asynccnt)
#define USE_ASYNC_LDS 1
#else
#define USE_ASYNC_LDS 0
#endif

#if USE_ASYNC_LDS
typedef __attribute__((address_space(1))) v4i_vec GVec;   // global (AS1) v4i
typedef __attribute__((address_space(3))) v4i_vec LVec;   // LDS (AS3) v4i

__device__ __forceinline__ void async_b128(const bf16_t* g, bf16_t* l) {
    // low 32 bits of a generic LDS pointer are the LDS offset
    __builtin_amdgcn_global_load_async_to_lds_b128(
        (GVec*)(unsigned long long)(uintptr_t)g,
        (LVec*)(unsigned)(uintptr_t)l,
        0, 0);
}
#endif

// ---------------------------------------------------------------- utilities
__device__ __forceinline__ unsigned long long ckey64(float v, int idx) {
    unsigned u = __float_as_uint(v);
    unsigned k = (u & 0x80000000u) ? ~u : (u | 0x80000000u);   // order-preserving flip
    return ((unsigned long long)k << 32) | (unsigned)(~(unsigned)idx); // value desc, idx asc
}
__device__ __forceinline__ float ckey_val(unsigned long long key) {
    unsigned k = (unsigned)(key >> 32);
    unsigned u = (k & 0x80000000u) ? (k ^ 0x80000000u) : ~k;
    return __uint_as_float(u);
}
__device__ __forceinline__ int ckey_idx(unsigned long long key) {
    return (int)(~(unsigned)(key & 0xFFFFFFFFu));
}

// -------------------------------------------------------- conversion kernels
__global__ void cvt_center_x(const float* __restrict__ x, const float* __restrict__ pre_bias,
                             bf16_t* __restrict__ xc, int D, long n) {
    long id = (long)blockIdx.x * blockDim.x + threadIdx.x;
    if (id >= n) return;
    int d = (int)(id % D);
    xc[id] = (bf16_t)(x[id] - pre_bias[d]);
}

__global__ void cvt_bf16(const float* __restrict__ src, bf16_t* __restrict__ dst, long n) {
    long id = (long)blockIdx.x * blockDim.x + threadIdx.x;
    if (id >= n) return;
    dst[id] = (bf16_t)src[id];
}

// transpose dec_W [D,N] f32 -> decWt [N,D] bf16
__global__ void transpose_cvt(const float* __restrict__ W, bf16_t* __restrict__ Wt, int D, int N) {
    __shared__ float tile[32][33];
    int tx = threadIdx.x, ty = threadIdx.y;
    int nb = blockIdx.x * 32, db = blockIdx.y * 32;
    #pragma unroll
    for (int i = 0; i < 4; ++i)
        tile[ty + 8 * i][tx] = W[(size_t)(db + ty + 8 * i) * N + (nb + tx)];
    __syncthreads();
    #pragma unroll
    for (int i = 0; i < 4; ++i)
        Wt[(size_t)(nb + ty + 8 * i) * D + (db + tx)] = (bf16_t)tile[tx][ty + 8 * i];
}

// ------------------------------------------------------------- encoder GEMM
// C[b,n] = sum_k xc_bf[b,k] * encW_bf[n,k]  (+latent_bias), 128x128 tile, 8 waves,
// double-buffered LDS fed by GLOBAL_LOAD_ASYNC_TO_LDS_B128 (ASYNCcnt-tracked)
__global__ __launch_bounds__(256) void encoder_gemm(
        const bf16_t* __restrict__ A,   // [B, D] bf16 (centered x)
        const bf16_t* __restrict__ Bw,  // [N, D] bf16 (enc_W)
        const float*  __restrict__ lbias,
        float* __restrict__ out_pre, float* __restrict__ out_jump,
        int D, int N) {
    __shared__ bf16_t As[2][128 * 32];
    __shared__ bf16_t Bs[2][128 * 32];

    const int t    = threadIdx.x;
    const int lane = t & 31;
    const int wave = t >> 5;
    const int wm   = (wave >> 1) * 32;   // 4 waves along M
    const int wn   = (wave & 1) * 64;    // 2 waves along N
    const int ln15 = lane & 15;
    const int lh   = lane >> 4;          // K-half selector

    const int gmBase = blockIdx.y * 128;
    const int gnBase = blockIdx.x * 128;
    const int lrow = t >> 1;
    const int lseg = (t & 1) * 16;
    const bf16_t* gA = A  + (size_t)(gmBase + lrow) * D + lseg;
    const bf16_t* gB = Bw + (size_t)(gnBase + lrow) * D + lseg;

    v8f acc[2][4];
    #pragma unroll
    for (int mi = 0; mi < 2; ++mi)
        #pragma unroll
        for (int ni = 0; ni < 4; ++ni)
            #pragma unroll
            for (int e = 0; e < 8; ++e) acc[mi][ni][e] = 0.f;

    // each thread stages its 16 bf16 (32B) of A and of B for one tile
    auto issue_tile = [&](int buf, int k0) {
        bf16_t* la = &As[buf][lrow * 32 + lseg];
        bf16_t* lb = &Bs[buf][lrow * 32 + lseg];
#if USE_ASYNC_LDS
        async_b128(gA + k0,     la);
        async_b128(gA + k0 + 8, la + 8);
        async_b128(gB + k0,     lb);
        async_b128(gB + k0 + 8, lb + 8);
#else
        *(v16bf*)la = *(const v16bf*)(gA + k0);
        *(v16bf*)lb = *(const v16bf*)(gB + k0);
#endif
    };

    const int nt = D / 32;
    issue_tile(0, 0);

    for (int it = 0; it < nt; ++it) {
        const int buf = it & 1;
        if (it + 1 < nt) {
            issue_tile(buf ^ 1, (it + 1) * 32);   // prefetch next tile into other buffer
#if USE_ASYNC_LDS
            __builtin_amdgcn_s_wait_asynccnt(4);  // wait current tile only (4 ops in flight)
#endif
        } else {
#if USE_ASYNC_LDS
            __builtin_amdgcn_s_wait_asynccnt(0);
#endif
        }
        __syncthreads();

        const bf16_t* as = As[buf];
        const bf16_t* bs = Bs[buf];
        v16bf af[2];
        #pragma unroll
        for (int mi = 0; mi < 2; ++mi) {
            int m = wm + mi * 16 + ln15;
            v8bf lo = *(const v8bf*)&as[m * 32 + 8 * lh];        // K = 8*lh .. +7
            v8bf hi = *(const v8bf*)&as[m * 32 + 16 + 8 * lh];   // K = 16+8*lh .. +7
            af[mi] = __builtin_shufflevector(lo, hi, 0,1,2,3,4,5,6,7,8,9,10,11,12,13,14,15);
        }
        v16bf bfr[4];
        #pragma unroll
        for (int ni = 0; ni < 4; ++ni) {
            int n = wn + ni * 16 + ln15;
            v8bf blo = *(const v8bf*)&bs[n * 32 + 8 * lh];       // K = 8*lh .. +7
            v8bf bhi = *(const v8bf*)&bs[n * 32 + 16 + 8 * lh];  // K = 16+8*lh .. +7
            bfr[ni] = __builtin_shufflevector(blo, bhi, 0,1,2,3,4,5,6,7,8,9,10,11,12,13,14,15);
        }
        #pragma unroll
        for (int mi = 0; mi < 2; ++mi)
            #pragma unroll
            for (int ni = 0; ni < 4; ++ni)
                acc[mi][ni] = __builtin_amdgcn_wmma_f32_16x16x32_bf16(
                    false, af[mi], false, bfr[ni], (short)0, acc[mi][ni], false, false);
        __syncthreads();   // protect buffer reuse two iterations later
    }

    // epilogue: C layout — VGPR e, lanes 0-15: M=e, N=lane; lanes 16-31: M=e+8, N=lane-16
    #pragma unroll
    for (int ni = 0; ni < 4; ++ni) {
        int gn = gnBase + wn + ni * 16 + ln15;
        float lb = lbias[gn];
        #pragma unroll
        for (int mi = 0; mi < 2; ++mi) {
            int gm0 = gmBase + wm + mi * 16 + 8 * lh;
            #pragma unroll
            for (int e = 0; e < 8; ++e) {
                float v = acc[mi][ni][e] + lb;
                size_t o = (size_t)(gm0 + e) * N + gn;
                out_pre[o]  = v;
                out_jump[o] = (v < THETA) ? 0.f : v;
            }
        }
    }
}

// ------------------------------------------------ deterministic top-K select
// One block per row. 64-bit composite key radix-select (8x 8-bit digits) +
// compaction + bitonic sort. stats!=null => auxk dead-mask mode.
__global__ __launch_bounds__(256) void topk_select(
        const float* __restrict__ pre,      // [B, N]
        const int*   __restrict__ stats,    // nullable (auxk mode)
        float* __restrict__ out_v,          // [B, KSEL]
        float* __restrict__ out_i,          // [B, KSEL] (float-cast indices)
        float* __restrict__ tk_v,           // nullable [B, KTOP]
        float* __restrict__ tk_i,           // nullable
        float* __restrict__ ws_v,           // nullable [B, KSEL] (relu'd, for decoder)
        int*   __restrict__ ws_i,           // nullable
        int N) {
    __shared__ unsigned hist[256];
    __shared__ unsigned long long prefix_s;
    __shared__ int need_s;
    __shared__ unsigned cnt_s;
    __shared__ unsigned long long ck[KSEL];

    const int t = threadIdx.x;
    const float* row = pre + (size_t)blockIdx.x * N;

    if (t == 0) { prefix_s = 0ull; need_s = KSEL; cnt_s = 0; }
    __syncthreads();

    for (int p = 56; p >= 0; p -= 8) {
        hist[t] = 0;
        __syncthreads();
        unsigned long long pf = prefix_s;
        for (int i = t; i < N; i += 256) {
            float v = row[i];
            if (stats) v = (stats[i] > DEADTH) ? v : v * 0.0f;
            unsigned long long key = ckey64(v, i);
            bool ok = (p == 56) || ((key >> (p + 8)) == pf);
            if (ok) atomicAdd(&hist[(unsigned)(key >> p) & 255u], 1u);
        }
        __syncthreads();
        if (t == 0) {
            int rem = need_s;
            unsigned long long np = prefix_s;
            for (int d = 255; d >= 0; --d) {
                int c = (int)hist[d];
                if (c >= rem) { np = (np << 8) | (unsigned)d; break; }
                rem -= c;
            }
            prefix_s = np;
            need_s = rem;
        }
        __syncthreads();
    }
    const unsigned long long T = prefix_s;   // KSEL-th largest composite key

    for (int i = t; i < N; i += 256) {
        float v = row[i];
        if (stats) v = (stats[i] > DEADTH) ? v : v * 0.0f;
        unsigned long long key = ckey64(v, i);
        if (key >= T) {
            unsigned pos = atomicAdd(&cnt_s, 1u);
            if (pos < KSEL) ck[pos] = key;
        }
    }
    __syncthreads();

    // bitonic sort, descending (keys unique => deterministic)
    for (unsigned ks = 2; ks <= KSEL; ks <<= 1) {
        for (unsigned j = ks >> 1; j > 0; j >>= 1) {
            unsigned i = (unsigned)t, ixj = i ^ j;
            if (ixj > i) {
                unsigned long long a = ck[i], b = ck[ixj];
                bool lower = ((i & ks) == 0);
                if ((lower && a < b) || (!lower && a > b)) { ck[i] = b; ck[ixj] = a; }
            }
            __syncthreads();
        }
    }

    unsigned long long key = ck[t];
    int   idx = ckey_idx(key);
    float val = fmaxf(ckey_val(key), 0.f);   // relu
    size_t ob = (size_t)blockIdx.x * KSEL + t;
    out_v[ob] = val;
    out_i[ob] = (float)idx;
    if (ws_v) { ws_v[ob] = val; ws_i[ob] = idx; }
    if (tk_v && t < KTOP) {
        size_t tb = (size_t)blockIdx.x * KTOP + t;
        tk_v[tb] = val;
        tk_i[tb] = (float)idx;
    }
}

// ----------------------------------------------------------------- stats ops
__global__ void stats_inc(const int* __restrict__ s_in, int* __restrict__ s_ws, int N) {
    int n = blockIdx.x * blockDim.x + threadIdx.x;
    if (n < N) s_ws[n] = s_in[n] + 1;
}
__global__ void stats_zero(const int* __restrict__ mk_idx, int* __restrict__ s_ws, int n) {
    int id = blockIdx.x * blockDim.x + threadIdx.x;
    if (id >= n) return;
    int r = id >> 6, j = id & 63;                 // top-64 are the first 64 of multik
    s_ws[mk_idx[r * KSEL + j]] = 0;               // all writes are 0 -> race-free
}
__global__ void stats_out(const int* __restrict__ s_ws, float* __restrict__ out, int N) {
    int n = blockIdx.x * blockDim.x + threadIdx.x;
    if (n < N) out[n] = (float)s_ws[n];
}

// ---------------------------------------------------------- sparse decoder
// recons from first 64 (== top-64 since multik is sorted), multik from all 256
__global__ __launch_bounds__(256) void sparse_decoder(
        const float* __restrict__ vals, const int* __restrict__ idxs,  // [B,256]
        const bf16_t* __restrict__ Wt,                                  // [N,D] bf16
        const float* __restrict__ pre_bias,
        float* __restrict__ recons, float* __restrict__ mrecons,
        int D, int N) {
    __shared__ float vs[KSEL];
    __shared__ int   is[KSEL];
    const int b = blockIdx.x, t = threadIdx.x;
    vs[t] = vals[(size_t)b * KSEL + t];
    is[t] = idxs[(size_t)b * KSEL + t];
    __syncthreads();

    for (int d0 = t * 8; d0 < D; d0 += 256 * 8) {
        float acc[8];
        #pragma unroll
        for (int i = 0; i < 8; ++i) acc[i] = pre_bias[d0 + i];

        for (int j = 0; j < KTOP; ++j) {
            float v = vs[j];
            const v8bf w = *(const v8bf*)(Wt + (size_t)is[j] * D + d0);
            if (j + 1 < KSEL) __builtin_prefetch(Wt + (size_t)is[j + 1] * D + d0, 0, 1);
            #pragma unroll
            for (int i = 0; i < 8; ++i) acc[i] += v * (float)w[i];
        }
        #pragma unroll
        for (int i = 0; i < 8; ++i) recons[(size_t)b * D + d0 + i] = acc[i];

        for (int j = KTOP; j < KSEL; ++j) {
            float v = vs[j];
            const v8bf w = *(const v8bf*)(Wt + (size_t)is[j] * D + d0);
            #pragma unroll
            for (int i = 0; i < 8; ++i) acc[i] += v * (float)w[i];
        }
        #pragma unroll
        for (int i = 0; i < 8; ++i) mrecons[(size_t)b * D + d0 + i] = acc[i];
    }
}

// ------------------------------------------------------------------- launch
extern "C" void kernel_launch(void* const* d_in, const int* in_sizes, int n_in,
                              void* d_out, int out_size, void* d_ws, size_t ws_size,
                              hipStream_t stream) {
    const float* x        = (const float*)d_in[0];
    const float* encW     = (const float*)d_in[1];
    const float* decW     = (const float*)d_in[2];
    const float* pre_bias = (const float*)d_in[3];
    const float* lbias    = (const float*)d_in[4];
    const int*   stats    = (const int*)d_in[5];

    const int D = in_sizes[3];
    const int N = in_sizes[4];
    const int B = in_sizes[0] / D;

    float* out = (float*)d_out;
    size_t o_recons   = 0;
    size_t o_mrecons  = o_recons  + (size_t)B * D;
    size_t o_tkv      = o_mrecons + (size_t)B * D;
    size_t o_tki      = o_tkv     + (size_t)B * KTOP;
    size_t o_mkv      = o_tki     + (size_t)B * KTOP;
    size_t o_mki      = o_mkv     + (size_t)B * KSEL;
    size_t o_akv      = o_mki     + (size_t)B * KSEL;
    size_t o_aki      = o_akv     + (size_t)B * KSEL;
    size_t o_pre      = o_aki     + (size_t)B * KSEL;
    size_t o_jump     = o_pre     + (size_t)B * N;
    size_t o_stats    = o_jump    + (size_t)B * N;

    // workspace carve (256B aligned)
    char* w = (char*)d_ws;
    auto carve = [&](size_t bytes) {
        char* p = w;
        w += (bytes + 255) & ~(size_t)255;
        return p;
    };
    bf16_t* xc_bf   = (bf16_t*)carve((size_t)B * D * sizeof(bf16_t));
    bf16_t* encW_bf = (bf16_t*)carve((size_t)N * D * sizeof(bf16_t));
    bf16_t* decWt   = (bf16_t*)carve((size_t)N * D * sizeof(bf16_t));
    float*  mkv_ws  = (float*)carve((size_t)B * KSEL * sizeof(float));
    int*    mki_ws  = (int*)carve((size_t)B * KSEL * sizeof(int));
    int*    st_ws   = (int*)carve((size_t)N * sizeof(int));

    long nx = (long)B * D, nw = (long)N * D;
    cvt_center_x<<<dim3((unsigned)((nx + 255) / 256)), 256, 0, stream>>>(x, pre_bias, xc_bf, D, nx);
    cvt_bf16    <<<dim3((unsigned)((nw + 255) / 256)), 256, 0, stream>>>(encW, encW_bf, nw);
    transpose_cvt<<<dim3(N / 32, D / 32), dim3(32, 8), 0, stream>>>(decW, decWt, D, N);

    encoder_gemm<<<dim3(N / 128, B / 128), 256, 0, stream>>>(
        xc_bf, encW_bf, lbias, out + o_pre, out + o_jump, D, N);

    topk_select<<<B, 256, 0, stream>>>(
        out + o_pre, (const int*)nullptr,
        out + o_mkv, out + o_mki, out + o_tkv, out + o_tki, mkv_ws, mki_ws, N);

    stats_inc <<<(N + 255) / 256, 256, 0, stream>>>(stats, st_ws, N);
    stats_zero<<<(B * KTOP + 255) / 256, 256, 0, stream>>>(mki_ws, st_ws, B * KTOP);
    stats_out <<<(N + 255) / 256, 256, 0, stream>>>(st_ws, out + o_stats, N);

    topk_select<<<B, 256, 0, stream>>>(
        out + o_pre, st_ws,
        out + o_akv, out + o_aki, (float*)nullptr, (float*)nullptr,
        (float*)nullptr, (int*)nullptr, N);

    sparse_decoder<<<B, 256, 0, stream>>>(
        mkv_ws, mki_ws, decWt, pre_bias, out + o_recons, out + o_mrecons, D, N);

    (void)n_in; (void)out_size; (void)ws_size;
}
